// GraphEmbedder_83811991814272
// MI455X (gfx1250) — matrix operand order repaired and checked by reading, not compiled, not run on hardware
//
#include <hip/hip_runtime.h>

#define N_NODES  100000
#define N_EDGES  300000
#define N_TYPES  3
#define HID      128
#define EMB      64
#define N_GRAPHS 4096
#define T_STEPS  4

typedef __attribute__((ext_vector_type(16))) __bf16 v16bf;
typedef __attribute__((ext_vector_type(8)))  float  v8f;

// A operand (16x32 bf16 tile) from an f32 row-major matrix with row stride HID.
// Lane L: row = base_row + (L&15); holds k = kb..kb+15, kb = k0 + (L>=16)*16.
// Native f32->bf16 converts (v_cvt_pk_bf16_f32 class, RNE).
__device__ __forceinline__ v16bf load_a_f32(const float* rowptr, int k0, int lane) {
    const int kA = k0 + ((lane & 16) ? 16 : 0);
    v8f lo = *(const v8f*)(rowptr + kA);
    v8f hi = *(const v8f*)(rowptr + kA + 8);
    v16bf a;
#pragma unroll
    for (int j = 0; j < 8; ++j) {
        a[j]     = (__bf16)lo[j];
        a[j + 8] = (__bf16)hi[j];
    }
    return a;
}

// B operand (32x16 bf16 tile) from column-major bf16 weights Bt[n][k] (k stride 1, n stride HID).
// Lane L: col n passed in; holds k = k0 + (L>=16)*16 + j, j=0..15 (contiguous 32B).
__device__ __forceinline__ v16bf load_b_bf16(const __bf16* Bt, int n, int k0, int lane) {
    const __bf16* p = Bt + (size_t)n * HID + k0 + ((lane & 16) ? 16 : 0);
    return *(const v16bf*)p;
}

__device__ __forceinline__ v8f wmma_bf16(v16bf a, v16bf b, v8f c) {
    return __builtin_amdgcn_wmma_f32_16x16x32_bf16(false, a, false, b, (short)0, c, false, false);
}

__device__ __forceinline__ float sigmoidf_(float x) { return 1.f / (1.f + __expf(-x)); }

// ---------------------------------------------------------------- utilities
__global__ void zero_f32_kernel(float4* p, int count4) {
    int i = blockIdx.x * blockDim.x + threadIdx.x;
    if (i < count4) p[i] = make_float4(0.f, 0.f, 0.f, 0.f);
}

__global__ void copy_f32_kernel(const float4* in, float4* out, int count4) {
    int i = blockIdx.x * blockDim.x + threadIdx.x;
    if (i < count4) out[i] = in[i];
}

__global__ void cvt_bf16_kernel(const float* in, __bf16* out, int count) {
    int i = blockIdx.x * blockDim.x + threadIdx.x;
    if (i < count) out[i] = (__bf16)in[i];
}

// W_msg[t][k][n]  ->  Bt[t][n][k]  (bf16, column-major for WMMA B operand)
__global__ void cvt_bf16_t_kernel(const float* W, __bf16* Bt) {
    int i = blockIdx.x * blockDim.x + threadIdx.x;
    if (i >= N_TYPES * HID * HID) return;
    int t = i / (HID * HID);
    int r = i % (HID * HID);
    int n = r / HID, k = r % HID;
    Bt[i] = (__bf16)W[t * HID * HID + k * HID + n];
}

// ---------------------------------------------------------------- edge scatter
// One 32-lane group per edge; lane handles 4 contiguous floats of the 128-wide row.
__global__ void scatter_kernel(const int* __restrict__ ei, int t,
                               const float* __restrict__ h,
                               float* __restrict__ Abuf,
                               float* __restrict__ deg) {
    int g = blockIdx.x * blockDim.x + threadIdx.x;
    int e = g >> 5;
    int lane = g & 31;
    if (e >= N_EDGES) return;
    const size_t base = (size_t)t * 2 * N_EDGES;
    int src = ei[base + e];
    int dst = ei[base + N_EDGES + e];
    const float4 v = *(const float4*)(h + (size_t)src * HID + lane * 4);
    float* dp = Abuf + (size_t)dst * HID + lane * 4;
    atomicAdd(dp + 0, v.x);
    atomicAdd(dp + 1, v.y);
    atomicAdd(dp + 2, v.z);
    atomicAdd(dp + 3, v.w);
    if (lane == 0) atomicAdd(&deg[dst], 1.0f);
}

// ---------------------------------------------------------------- message GEMM
// m[N,128] += Abuf[N,128] @ W_t  (+ deg * b_t).  8 waves/block, 16 rows/wave, full 128 cols.
__global__ void gemm_msg_kernel(const float* __restrict__ Abuf,
                                const __bf16* __restrict__ Bt,
                                const float* __restrict__ deg,
                                const float* __restrict__ bias,
                                float* __restrict__ m) {
    int tid = threadIdx.x, wave = tid >> 5, lane = tid & 31;
    int r0 = blockIdx.x * 128 + wave * 16;
    int rowA = r0 + (lane & 15);
    int rowAc = rowA < N_NODES ? rowA : N_NODES - 1;
    const float* arow = Abuf + (size_t)rowAc * HID;

    v8f acc[8] = {};
#pragma unroll
    for (int k0 = 0; k0 < HID; k0 += 32) {
        v16bf a = load_a_f32(arow, k0, lane);
#pragma unroll
        for (int nt = 0; nt < 8; ++nt) {
            v16bf b = load_b_bf16(Bt, nt * 16 + (lane & 15), k0, lane);
            acc[nt] = wmma_bf16(a, b, acc[nt]);
        }
    }

    const int rhi = (lane & 16) ? 8 : 0;
    if (r0 + 16 <= N_NODES) {            // wave-uniform: fast unmasked path
        float degv[8];
#pragma unroll
        for (int r = 0; r < 8; ++r) degv[r] = deg[r0 + rhi + r];
#pragma unroll
        for (int nt = 0; nt < 8; ++nt) {
            int col = nt * 16 + (lane & 15);
            float bv = bias[col];
#pragma unroll
            for (int r = 0; r < 8; ++r) {
                size_t idx = (size_t)(r0 + rhi + r) * HID + col;
                m[idx] += acc[nt][r] + degv[r] * bv;
            }
        }
    } else {                              // only the last block takes this
        float degv[8];
#pragma unroll
        for (int r = 0; r < 8; ++r) {
            int row = r0 + rhi + r;
            degv[r] = (row < N_NODES) ? deg[row] : 0.f;
        }
#pragma unroll
        for (int nt = 0; nt < 8; ++nt) {
            int col = nt * 16 + (lane & 15);
            float bv = bias[col];
#pragma unroll
            for (int r = 0; r < 8; ++r) {
                int row = r0 + rhi + r;
                if (row < N_NODES) {
                    size_t idx = (size_t)row * HID + col;
                    m[idx] += acc[nt][r] + degv[r] * bv;
                }
            }
        }
    }
}

// ---------------------------------------------------------------- fused GRU step
// Per block: one 16-node tile. gi = m@W_ih^T, gh = h@W_hh^T into LDS, then gate math, h in place.
// 8 waves; wave w owns gate columns [w*48, w*48+48) for both gi and gh.
__global__ void gru_kernel(const float* __restrict__ m,
                           float* __restrict__ h,
                           const __bf16* __restrict__ Wih,
                           const __bf16* __restrict__ Whh,
                           const float* __restrict__ b_ih,
                           const float* __restrict__ b_hh) {
    __shared__ float gi[16 * 384];
    __shared__ float gh[16 * 384];

    int tid = threadIdx.x, wave = tid >> 5, lane = tid & 31;
    int rbase = blockIdx.x * 16;     // 100000 / 16 = 6250 exact, no masking needed

    const float* mrow = m + (size_t)(rbase + (lane & 15)) * HID;
    const float* hrow = h + (size_t)(rbase + (lane & 15)) * HID;

    v8f acci[3] = {};
    v8f acch[3] = {};
#pragma unroll
    for (int k0 = 0; k0 < HID; k0 += 32) {
        v16bf am = load_a_f32(mrow, k0, lane);
        v16bf ah = load_a_f32(hrow, k0, lane);
#pragma unroll
        for (int nt = 0; nt < 3; ++nt) {
            int g = wave * 48 + nt * 16 + (lane & 15);
            v16bf bi = load_b_bf16(Wih, g, k0, lane);
            v16bf bh = load_b_bf16(Whh, g, k0, lane);
            acci[nt] = wmma_bf16(am, bi, acci[nt]);
            acch[nt] = wmma_bf16(ah, bh, acch[nt]);
        }
    }

    const int rhi = (lane & 16) ? 8 : 0;
#pragma unroll
    for (int nt = 0; nt < 3; ++nt) {
        int col = wave * 48 + nt * 16 + (lane & 15);
        float bi = b_ih[col], bh = b_hh[col];
#pragma unroll
        for (int r = 0; r < 8; ++r) {
            int row = r + rhi;
            gi[row * 384 + col] = acci[nt][r] + bi;
            gh[row * 384 + col] = acch[nt][r] + bh;
        }
    }
    __syncthreads();

    for (int e = tid; e < 16 * HID; e += 256) {
        int i = e >> 7, d = e & 127;
        float r  = sigmoidf_(gi[i * 384 + d]       + gh[i * 384 + d]);
        float z  = sigmoidf_(gi[i * 384 + 128 + d] + gh[i * 384 + 128 + d]);
        float nn = tanhf(gi[i * 384 + 256 + d] + r * gh[i * 384 + 256 + d]);
        size_t idx = (size_t)(rbase + i) * HID + d;
        float hv = h[idx];
        h[idx] = (1.f - z) * nn + z * hv;
    }
}

// ---------------------------------------------------------------- gated readout
// proj = h@W_proj^T + b ; gate = sigmoid(h@W_gate^T + b) ; atomicAdd into out[graph][col].
__global__ void readout_kernel(const float* __restrict__ h,
                               const __bf16* __restrict__ Wp,
                               const float* __restrict__ bp,
                               const __bf16* __restrict__ Wg,
                               const float* __restrict__ bg,
                               const int* __restrict__ n2g,
                               float* __restrict__ out) {
    int tid = threadIdx.x, wave = tid >> 5, lane = tid & 31;
    int r0 = blockIdx.x * 128 + wave * 16;
    int rowA = r0 + (lane & 15);
    int rowAc = rowA < N_NODES ? rowA : N_NODES - 1;
    const float* arow = h + (size_t)rowAc * HID;

    v8f ap[4] = {};
    v8f ag[4] = {};
#pragma unroll
    for (int k0 = 0; k0 < HID; k0 += 32) {
        v16bf a = load_a_f32(arow, k0, lane);
#pragma unroll
        for (int nt = 0; nt < 4; ++nt) {
            int n = nt * 16 + (lane & 15);
            v16bf bpv = load_b_bf16(Wp, n, k0, lane);
            v16bf bgv = load_b_bf16(Wg, n, k0, lane);
            ap[nt] = wmma_bf16(a, bpv, ap[nt]);
            ag[nt] = wmma_bf16(a, bgv, ag[nt]);
        }
    }

    const int rhi = (lane & 16) ? 8 : 0;
    float bpv[4], bgv[4];
#pragma unroll
    for (int nt = 0; nt < 4; ++nt) {
        int col = nt * 16 + (lane & 15);
        bpv[nt] = bp[col];
        bgv[nt] = bg[col];
    }

    if (r0 + 16 <= N_NODES) {            // wave-uniform fast path
        int gid[8];
#pragma unroll
        for (int r = 0; r < 8; ++r) gid[r] = n2g[r0 + rhi + r];
#pragma unroll
        for (int r = 0; r < 8; ++r) {
#pragma unroll
            for (int nt = 0; nt < 4; ++nt) {
                int col = nt * 16 + (lane & 15);
                float p = ap[nt][r] + bpv[nt];
                float g = sigmoidf_(ag[nt][r] + bgv[nt]);
                atomicAdd(&out[(size_t)gid[r] * EMB + col], p * g);
            }
        }
    } else {
#pragma unroll
        for (int r = 0; r < 8; ++r) {
            int row = r0 + rhi + r;
            if (row < N_NODES) {
                int gid = n2g[row];
#pragma unroll
                for (int nt = 0; nt < 4; ++nt) {
                    int col = nt * 16 + (lane & 15);
                    float p = ap[nt][r] + bpv[nt];
                    float g = sigmoidf_(ag[nt][r] + bgv[nt]);
                    atomicAdd(&out[(size_t)gid * EMB + col], p * g);
                }
            }
        }
    }
}

// ---------------------------------------------------------------- launch
extern "C" void kernel_launch(void* const* d_in, const int* in_sizes, int n_in,
                              void* d_out, int out_size, void* d_ws, size_t ws_size,
                              hipStream_t stream) {
    const float* node_features = (const float*)d_in[0];
    const int*   edge_index    = (const int*)d_in[1];
    const int*   n2g           = (const int*)d_in[2];
    const float* W_msg  = (const float*)d_in[3];
    const float* b_msg  = (const float*)d_in[4];
    const float* W_ih   = (const float*)d_in[5];
    const float* W_hh   = (const float*)d_in[6];
    const float* b_ih   = (const float*)d_in[7];
    const float* b_hh   = (const float*)d_in[8];
    const float* W_proj = (const float*)d_in[9];
    const float* b_proj = (const float*)d_in[10];
    const float* W_gate = (const float*)d_in[11];
    const float* b_gate = (const float*)d_in[12];
    float* out = (float*)d_out;

    char* ws = (char*)d_ws;
    constexpr size_t SZ_NH = (size_t)N_NODES * HID * sizeof(float);   // 51.2 MB
    float* h    = (float*)(ws);
    float* Abuf = (float*)(ws + SZ_NH);
    float* m    = (float*)(ws + 2 * SZ_NH);
    float* deg  = (float*)(ws + 3 * SZ_NH);
    size_t off = 3 * SZ_NH + (((size_t)N_NODES * sizeof(float) + 511) / 512) * 512;
    __bf16* Wmsg_bt = (__bf16*)(ws + off); off += (size_t)N_TYPES * HID * HID * 2;
    __bf16* Wih_bf  = (__bf16*)(ws + off); off += (size_t)3 * HID * HID * 2;
    __bf16* Whh_bf  = (__bf16*)(ws + off); off += (size_t)3 * HID * HID * 2;
    __bf16* Wp_bf   = (__bf16*)(ws + off); off += (size_t)EMB * HID * 2;
    __bf16* Wg_bf   = (__bf16*)(ws + off); off += (size_t)EMB * HID * 2;
    (void)ws_size; (void)in_sizes; (void)n_in; (void)out_size;

    const int NH4   = N_NODES * HID / 4;       // 3.2M float4
    const int DEG4  = N_NODES / 4;
    const int OUT4  = N_GRAPHS * EMB / 4;
    const dim3 B(256);

    // one-time (per launch) weight conversion: bf16, column-major for WMMA B operand
    cvt_bf16_t_kernel<<<(N_TYPES * HID * HID + 255) / 256, B, 0, stream>>>(W_msg, Wmsg_bt);
    cvt_bf16_kernel<<<(3 * HID * HID + 255) / 256, B, 0, stream>>>(W_ih, Wih_bf, 3 * HID * HID);
    cvt_bf16_kernel<<<(3 * HID * HID + 255) / 256, B, 0, stream>>>(W_hh, Whh_bf, 3 * HID * HID);
    cvt_bf16_kernel<<<(EMB * HID + 255) / 256, B, 0, stream>>>(W_proj, Wp_bf, EMB * HID);
    cvt_bf16_kernel<<<(EMB * HID + 255) / 256, B, 0, stream>>>(W_gate, Wg_bf, EMB * HID);

    // h = node_features (state kept in f32; inputs never mutated)
    copy_f32_kernel<<<(NH4 + 255) / 256, B, 0, stream>>>((const float4*)node_features, (float4*)h, NH4);
    // output accumulator must start at zero (harness poisons it)
    zero_f32_kernel<<<(OUT4 + 255) / 256, B, 0, stream>>>((float4*)out, OUT4);

    const int scatter_blocks = (N_EDGES * 32 + 255) / 256;   // 32 lanes per edge
    const int gemm_blocks    = (N_NODES + 127) / 128;        // 8 waves x 16 rows
    const int gru_blocks     = N_NODES / 16;                 // exact

    for (int step = 0; step < T_STEPS; ++step) {
        zero_f32_kernel<<<(NH4 + 255) / 256, B, 0, stream>>>((float4*)m, NH4);
        for (int t = 0; t < N_TYPES; ++t) {
            zero_f32_kernel<<<(NH4 + 255) / 256, B, 0, stream>>>((float4*)Abuf, NH4);
            zero_f32_kernel<<<(DEG4 + 255) / 256, B, 0, stream>>>((float4*)deg, DEG4);
            scatter_kernel<<<scatter_blocks, B, 0, stream>>>(edge_index, t, h, Abuf, deg);
            gemm_msg_kernel<<<gemm_blocks, B, 0, stream>>>(
                Abuf, Wmsg_bt + (size_t)t * HID * HID, deg, b_msg + (size_t)t * HID, m);
        }
        gru_kernel<<<gru_blocks, B, 0, stream>>>(m, h, Wih_bf, Whh_bf, b_ih, b_hh);
    }

    readout_kernel<<<gemm_blocks, B, 0, stream>>>(h, Wp_bf, b_proj, Wg_bf, b_gate, n2g, out);
}
